// GeometryMultiHeadAttention_52931176956436
// MI455X (gfx1250) — compile-verified, matrix operand order checked by reading, not run on hardware
//
#include <hip/hip_runtime.h>
#include <type_traits>

// ---------------------------------------------------------------------------
// GeometryMultiHeadAttention for MI455X (gfx1250, wave32, WMMA bf16)
// B=4, N=1024, D=512, H=8, HD=64, R=32
// ---------------------------------------------------------------------------

typedef __attribute__((ext_vector_type(16))) __bf16 v16bf;
typedef __attribute__((ext_vector_type(8)))  __bf16 v8bf;
typedef __attribute__((ext_vector_type(8)))  float  v8f;

#define B_  4
#define N_  1024
#define D_  512
#define H_  8
#define HD_ 64
#define R_  32

// Native conversions (bf16 is a first-class type on gfx1250)
__device__ __forceinline__ __bf16 f2bf(float f) { return (__bf16)f; }
__device__ __forceinline__ float  bf2f(__bf16 v) { return (float)v; }

__device__ __forceinline__ v16bf cat16(v8bf lo, v8bf hi) {
  return __builtin_shufflevector(lo, hi, 0,1,2,3,4,5,6,7,8,9,10,11,12,13,14,15);
}

__device__ __forceinline__ v8f wmma_bf16(v16bf a, v16bf b, v8f c) {
  // D = A(16x32 bf16) * B(32x16 bf16) + C(16x16 f32)
  return __builtin_amdgcn_wmma_f32_16x16x32_bf16(
      /*neg_a=*/false, a, /*neg_b=*/false, b,
      /*c_mod=*/(short)0, c, /*reuse_a=*/false, /*reuse_b=*/false);
}

// Async global->LDS copy of 32B (2 x b128). INST_OFFSET is added to both the
// LDS and the global address, so one base covers the strip.
__device__ __forceinline__ void async_copy32(void* lds, const void* gptr) {
  unsigned ldsa = (unsigned)(uintptr_t)lds;             // low 32 bits = LDS addr
  unsigned long long ga = (unsigned long long)(uintptr_t)gptr;
  asm volatile(
      "global_load_async_to_lds_b128 %0, %1, off\n\t"
      "global_load_async_to_lds_b128 %0, %1, off offset:16"
      :: "v"(ldsa), "v"(ga) : "memory");
}
__device__ __forceinline__ void async_copy64(void* lds, const void* gptr) {
  unsigned ldsa = (unsigned)(uintptr_t)lds;
  unsigned long long ga = (unsigned long long)(uintptr_t)gptr;
  asm volatile(
      "global_load_async_to_lds_b128 %0, %1, off\n\t"
      "global_load_async_to_lds_b128 %0, %1, off offset:16\n\t"
      "global_load_async_to_lds_b128 %0, %1, off offset:32\n\t"
      "global_load_async_to_lds_b128 %0, %1, off offset:48"
      :: "v"(ldsa), "v"(ga) : "memory");
}
__device__ __forceinline__ void wait_async() {
  asm volatile("s_wait_asynccnt 0x0" ::: "memory");
}

// ---------------------------------------------------------------------------
// f32 -> bf16 conversion
// ---------------------------------------------------------------------------
__global__ void cvt_f32_to_bf16(const float* __restrict__ src,
                                __bf16* __restrict__ dst, int n) {
  int i = blockIdx.x * blockDim.x + threadIdx.x;
  if (i < n) dst[i] = f2bf(src[i]);
}

// ---------------------------------------------------------------------------
// RBF bias precompute on the matrix cores.
// bias[b][h][n][m] = sum_r exp(-gamma (d(n,m)-c_r)^2) * rbf_w[r][h] + rbf_b[h]
// Each lane computes its 16 RBF features directly in WMMA A-fragment slots;
// one v_wmma_f32_16x16x32_bf16 projects 16 pairs x 8 heads at once.
// Grid: (8, N, B), block 256 (8 waves -> 8 m-tiles of 16).
// ---------------------------------------------------------------------------
__global__ __launch_bounds__(256) void rbf_bias_kernel(
    const float* __restrict__ pos, const float* __restrict__ rbf_w,
    const float* __restrict__ rbf_b, __bf16* __restrict__ biasT) {
  const float SPACING = 10.0f / 31.0f;
  const float GAMMA   = 1.0f / (SPACING * SPACING);
  const float GL      = GAMMA * 1.4426950408889634f;  // exp(-g t^2)=exp2(-GL t^2)

  const int tid = threadIdx.x;
  const int wv = tid >> 5, lane = tid & 31;
  const int lgrp = lane >> 4, l16 = lane & 15;
  const int n = blockIdx.y, b = blockIdx.z;
  const int m0 = (blockIdx.x * 8 + wv) * 16;
  const int rbase = lgrp * 8;  // K-slot map: e<8 -> rbase+e, else rbase+16+e-8

  v16bf wf;
#pragma unroll
  for (int e = 0; e < 16; e++) {
    int r = (e < 8) ? rbase + e : rbase + 16 + (e - 8);
    float w = (l16 < H_) ? rbf_w[r * H_ + l16] : 0.0f;
    wf[e] = f2bf(w);
  }
  float rbv = (l16 < H_) ? rbf_b[l16] : 0.0f;

  const float* pn = pos + (size_t)(b * N_ + n) * 3;
  float qx = pn[0], qy = pn[1], qz = pn[2];
  const int m = m0 + l16;
  const float* pm = pos + (size_t)(b * N_ + m) * 3;
  float dx = qx - pm[0], dy = qy - pm[1], dz = qz - pm[2];
  float d = __builtin_amdgcn_sqrtf(dx * dx + dy * dy + dz * dz);

  v16bf af;
#pragma unroll
  for (int e = 0; e < 16; e++) {
    int r = (e < 8) ? rbase + e : rbase + 16 + (e - 8);
    float t = d - (float)r * SPACING;
    af[e] = f2bf(__builtin_amdgcn_exp2f(-GL * t * t));
  }

  v8f c;
#pragma unroll
  for (int v = 0; v < 8; v++) c[v] = rbv;
  c = wmma_bf16(af, wf, c);

  if (l16 < H_) {
    v8bf outv;
#pragma unroll
    for (int v = 0; v < 8; v++) outv[v] = f2bf(c[v]);
    __bf16* dst = biasT + (((size_t)(b * H_ + l16) * N_ + n) * N_) + m0 + 8 * lgrp;
    *(v8bf*)dst = outv;
  }
}

// ---------------------------------------------------------------------------
// C[M x 512] = (A[M x 512] @ W[512 x 512] + bias) * scale
// Block: 256 threads (8 waves). Block tile 128x128, wave tile 64x32.
// A tile staged via global_load_async_to_lds; W tile transposed in registers.
// ---------------------------------------------------------------------------
template <typename OutT>
__global__ __launch_bounds__(256) void gemm_bf16_512(
    const __bf16* __restrict__ A, const __bf16* __restrict__ W,
    const float* __restrict__ bias, OutT* __restrict__ C, float scale) {
  constexpr int K = 512, NN = 512, KT = 32;
  __shared__ __align__(16) __bf16 As[128][KT + 8];  // row-major A tile
  __shared__ __align__(16) __bf16 Bt[128][KT + 8];  // transposed W tile: Bt[n][k]

  const int m0 = blockIdx.x * 128;
  const int n0 = blockIdx.y * 128;
  const int tid = threadIdx.x;
  const int wave = tid >> 5;
  const int lane = tid & 31;
  const int lgrp = lane >> 4;
  const int l16  = lane & 15;
  const int wm = (wave >> 2) * 64;
  const int wn = (wave & 3) * 32;

  v8f acc[4][2] = {};

  for (int k0 = 0; k0 < K; k0 += KT) {
    __syncthreads();
    {
      // A tile: straight copy -> async global->LDS (32B per thread)
      int row = tid >> 1, kc = (tid & 1) * 16;
      async_copy32(&As[row][kc], A + (size_t)(m0 + row) * K + k0 + kc);
      // W tile transposed into Bt
      int kk = tid & 31;
      int nc = (tid >> 5) * 16;
      const __bf16* wp = W + (size_t)(k0 + kk) * NN + n0 + nc;
      v8bf w0 = *(const v8bf*)wp;
      v8bf w1 = *(const v8bf*)(wp + 8);
#pragma unroll
      for (int j = 0; j < 8; j++) {
        Bt[nc + j][kk]     = w0[j];
        Bt[nc + 8 + j][kk] = w1[j];
      }
    }
    wait_async();
    __syncthreads();

    const int kc = lgrp * 8;
    v16bf af[4], bfrag[2];
#pragma unroll
    for (int rt = 0; rt < 4; rt++) {
      v8bf lo = *(const v8bf*)&As[wm + rt * 16 + l16][kc];
      v8bf hi = *(const v8bf*)&As[wm + rt * 16 + l16][kc + 16];
      af[rt] = cat16(lo, hi);
    }
#pragma unroll
    for (int ct = 0; ct < 2; ct++) {
      v8bf lo = *(const v8bf*)&Bt[wn + ct * 16 + l16][kc];
      v8bf hi = *(const v8bf*)&Bt[wn + ct * 16 + l16][kc + 16];
      bfrag[ct] = cat16(lo, hi);
    }
#pragma unroll
    for (int rt = 0; rt < 4; rt++)
#pragma unroll
      for (int ct = 0; ct < 2; ct++)
        acc[rt][ct] = wmma_bf16(af[rt], bfrag[ct], acc[rt][ct]);
  }

#pragma unroll
  for (int rt = 0; rt < 4; rt++) {
#pragma unroll
    for (int ct = 0; ct < 2; ct++) {
      int col = n0 + wn + ct * 16 + l16;
      float bv = bias[col];
#pragma unroll
      for (int v = 0; v < 8; v++) {
        int row = m0 + wm + rt * 16 + v + 8 * lgrp;
        float val = (acc[rt][ct][v] + bv) * scale;
        if constexpr (sizeof(OutT) == 2)
          C[(size_t)row * NN + col] = f2bf(val);
        else
          C[(size_t)row * NN + col] = val;
      }
    }
  }
}

// ---------------------------------------------------------------------------
// Fused flash attention. FUSED=true computes RBF bias inline (fallback when
// workspace is small); FUSED=false reads the precomputed bf16 bias tensor.
// K tile staged via async global->LDS; V transposed in registers.
// Grid: (N/64, H, B). Block: 128 threads (4 waves), each wave = 16 query rows.
// ---------------------------------------------------------------------------
template <bool FUSED>
__global__ __launch_bounds__(128) void attn_kernel(
    const __bf16* __restrict__ Qb, const __bf16* __restrict__ Kb,
    const __bf16* __restrict__ Vb, const float* __restrict__ pos,
    const int* __restrict__ mask, const float* __restrict__ rbf_w,
    const float* __restrict__ rbf_b, const __bf16* __restrict__ biasT,
    __bf16* __restrict__ Ob) {
  const float SPACING = 10.0f / 31.0f;
  const float GAMMA   = 1.0f / (SPACING * SPACING);
  const float LOG2E   = 1.4426950408889634f;
  const float GL      = GAMMA * LOG2E;

  __shared__ __align__(16) __bf16 Ks[64][72];
  __shared__ __align__(16) __bf16 Vt[64][72];
  __shared__ __align__(16) __bf16 Ps[4][16][72];
  __shared__ float kpos[64][3];
  __shared__ int   kmask[64];

  const int qt  = blockIdx.x;
  const int h   = blockIdx.y;
  const int b   = blockIdx.z;
  const int tid = threadIdx.x;
  const int wv   = tid >> 5;
  const int lane = tid & 31;
  const int lgrp = lane >> 4;
  const int l16  = lane & 15;
  const int qbase = qt * 64 + wv * 16;

  float wr[R_];
  float rb = 0.0f;
  if constexpr (FUSED) {
#pragma unroll
    for (int r = 0; r < R_; r++) wr[r] = rbf_w[r * H_ + h];
    rb = rbf_b[h];
  }

  v16bf qf[2];
  {
    const __bf16* qrow = Qb + (size_t)(b * N_ + qbase + l16) * D_ + h * HD_;
#pragma unroll
    for (int kk = 0; kk < 2; kk++) {
      int c0 = kk * 32 + lgrp * 8;
      v8bf lo = *(const v8bf*)(qrow + c0);
      v8bf hi = *(const v8bf*)(qrow + c0 + 16);
      qf[kk] = cat16(lo, hi);
    }
  }
  float qpx[8], qpy[8], qpz[8];
  if constexpr (FUSED) {
#pragma unroll
    for (int v = 0; v < 8; v++) {
      int row = qbase + v + 8 * lgrp;
      const float* p = pos + (size_t)(b * N_ + row) * 3;
      qpx[v] = p[0]; qpy[v] = p[1]; qpz[v] = p[2];
    }
  }
  const __bf16* bplane = biasT + ((size_t)(b * H_ + h) * N_) * N_;

  v8f oacc[4] = {};
  float mi[8], li[8];
#pragma unroll
  for (int v = 0; v < 8; v++) { mi[v] = -1e30f; li[v] = 0.0f; }

  for (int kb = 0; kb < N_; kb += 64) {
    __syncthreads();
    {
      int row = tid >> 1, c0 = (tid & 1) * 32;
      const __bf16* kg = Kb + (size_t)(b * N_ + kb + row) * D_ + h * HD_ + c0;
      const __bf16* vg = Vb + (size_t)(b * N_ + kb + row) * D_ + h * HD_ + c0;
      if (kb + 64 < N_) {  // prefetch next key block (global_prefetch_b8)
        __builtin_prefetch(kg + (size_t)64 * D_, 0, 1);
        __builtin_prefetch(vg + (size_t)64 * D_, 0, 1);
      }
      // K tile: straight copy -> async global->LDS (64B per thread)
      async_copy64(&Ks[row][c0], kg);
      // V tile: transpose in registers
      v8bf a0 = ((const v8bf*)vg)[0], a1 = ((const v8bf*)vg)[1];
      v8bf a2 = ((const v8bf*)vg)[2], a3 = ((const v8bf*)vg)[3];
#pragma unroll
      for (int j = 0; j < 8; j++) {
        Vt[c0 + j][row]      = a0[j];
        Vt[c0 + 8 + j][row]  = a1[j];
        Vt[c0 + 16 + j][row] = a2[j];
        Vt[c0 + 24 + j][row] = a3[j];
      }
      if (tid < 64) {
        kmask[tid] = mask[b * N_ + kb + tid];
        if constexpr (FUSED) {
          const float* pp = pos + (size_t)(b * N_ + kb + tid) * 3;
          kpos[tid][0] = pp[0]; kpos[tid][1] = pp[1]; kpos[tid][2] = pp[2];
        }
      }
    }
    wait_async();
    __syncthreads();

    // S = Q K^T (+ bias), in log2 domain
    float sc[4][8];
#pragma unroll
    for (int ct = 0; ct < 4; ct++) {
      v8f s = {};
#pragma unroll
      for (int kk = 0; kk < 2; kk++) {
        int c0 = kk * 32 + lgrp * 8;
        v8bf lo = *(const v8bf*)&Ks[ct * 16 + l16][c0];
        v8bf hi = *(const v8bf*)&Ks[ct * 16 + l16][c0 + 16];
        s = wmma_bf16(qf[kk], cat16(lo, hi), s);
      }
      int nk = ct * 16 + l16;
      bool ok = (kmask[nk] != 0);
      if constexpr (FUSED) {
        float kx = kpos[nk][0], ky = kpos[nk][1], kz = kpos[nk][2];
#pragma unroll
        for (int v = 0; v < 8; v++) {
          float dx = qpx[v] - kx, dy = qpy[v] - ky, dz = qpz[v] - kz;
          float d = __builtin_amdgcn_sqrtf(dx * dx + dy * dy + dz * dz);
          float bias = rb;
#pragma unroll
          for (int r = 0; r < R_; r++) {
            float t = d - (float)r * SPACING;
            bias = __builtin_fmaf(__builtin_amdgcn_exp2f(-GL * t * t), wr[r], bias);
          }
          float val = (s[v] + bias) * LOG2E;
          sc[ct][v] = ok ? val : -1e30f;
        }
      } else {
#pragma unroll
        for (int v = 0; v < 8; v++) {
          int row = qbase + v + 8 * lgrp;
          float bias = bf2f(bplane[(size_t)row * N_ + kb + nk]);
          float val = (s[v] + bias) * LOG2E;
          sc[ct][v] = ok ? val : -1e30f;
        }
      }
    }

    // Online softmax (row stats reduced across 16-lane groups)
    float alpha[8], mnew[8];
#pragma unroll
    for (int v = 0; v < 8; v++) {
      float mx = fmaxf(fmaxf(sc[0][v], sc[1][v]), fmaxf(sc[2][v], sc[3][v]));
#pragma unroll
      for (int off = 1; off < 16; off <<= 1)
        mx = fmaxf(mx, __shfl_xor(mx, off, 32));
      mnew[v] = fmaxf(mi[v], mx);
      alpha[v] = __builtin_amdgcn_exp2f(mi[v] - mnew[v]);
    }
#pragma unroll
    for (int ct = 0; ct < 4; ct++)
#pragma unroll
      for (int v = 0; v < 8; v++)
        sc[ct][v] = __builtin_amdgcn_exp2f(sc[ct][v] - mnew[v]);
#pragma unroll
    for (int v = 0; v < 8; v++) {
      float rs = (sc[0][v] + sc[1][v]) + (sc[2][v] + sc[3][v]);
#pragma unroll
      for (int off = 1; off < 16; off <<= 1) rs += __shfl_xor(rs, off, 32);
      li[v] = li[v] * alpha[v] + rs;
      mi[v] = mnew[v];
    }
#pragma unroll
    for (int ot = 0; ot < 4; ot++)
#pragma unroll
      for (int v = 0; v < 8; v++) oacc[ot][v] *= alpha[v];

    // P (C layout) -> wave-private LDS -> A layout fragments
#pragma unroll
    for (int ct = 0; ct < 4; ct++)
#pragma unroll
      for (int v = 0; v < 8; v++)
        Ps[wv][v + 8 * lgrp][ct * 16 + l16] = f2bf(sc[ct][v]);
    asm volatile("s_wait_dscnt 0" ::: "memory");

    v16bf pf[2];
#pragma unroll
    for (int kk = 0; kk < 2; kk++) {
      int c0 = kk * 32 + lgrp * 8;
      v8bf lo = *(const v8bf*)&Ps[wv][l16][c0];
      v8bf hi = *(const v8bf*)&Ps[wv][l16][c0 + 16];
      pf[kk] = cat16(lo, hi);
    }
#pragma unroll
    for (int ot = 0; ot < 4; ot++) {
#pragma unroll
      for (int kk = 0; kk < 2; kk++) {
        int c0 = kk * 32 + lgrp * 8;
        v8bf lo = *(const v8bf*)&Vt[ot * 16 + l16][c0];
        v8bf hi = *(const v8bf*)&Vt[ot * 16 + l16][c0 + 16];
        oacc[ot] = wmma_bf16(pf[kk], cat16(lo, hi), oacc[ot]);
      }
    }
  }

#pragma unroll
  for (int v = 0; v < 8; v++) {
    float inv = 1.0f / li[v];
    int row = qbase + v + 8 * lgrp;
#pragma unroll
    for (int ot = 0; ot < 4; ot++) {
      Ob[(size_t)(b * N_ + row) * D_ + h * HD_ + ot * 16 + l16] =
          f2bf(oacc[ot][v] * inv);
    }
  }
}

// ---------------------------------------------------------------------------
// Host launcher
// ---------------------------------------------------------------------------
extern "C" void kernel_launch(void* const* d_in, const int* in_sizes, int n_in,
                              void* d_out, int out_size, void* d_ws,
                              size_t ws_size, hipStream_t stream) {
  const float* h     = (const float*)d_in[0];
  const float* pos   = (const float*)d_in[1];
  const int*   mask  = (const int*)d_in[2];
  const float* q_w   = (const float*)d_in[3];
  const float* q_b   = (const float*)d_in[4];
  const float* k_w   = (const float*)d_in[5];
  const float* k_b   = (const float*)d_in[6];
  const float* v_w   = (const float*)d_in[7];
  const float* v_b   = (const float*)d_in[8];
  const float* o_w   = (const float*)d_in[9];
  const float* o_b   = (const float*)d_in[10];
  const float* rbf_w = (const float*)d_in[11];
  const float* rbf_b = (const float*)d_in[12];
  float* out = (float*)d_out;

  const size_t BND = (size_t)B_ * N_ * D_;  // 2,097,152
  const size_t DD  = (size_t)D_ * D_;       // 262,144

  char* ws = (char*)d_ws;
  size_t off = 0;
  auto carve = [&](size_t bytes) -> char* {
    char* p = ws + off;
    off += (bytes + 255) & ~(size_t)255;
    return p;
  };
  __bf16* hb  = (__bf16*)carve(BND * 2);
  __bf16* qwb = (__bf16*)carve(DD * 2);
  __bf16* kwb = (__bf16*)carve(DD * 2);
  __bf16* vwb = (__bf16*)carve(DD * 2);
  __bf16* owb = (__bf16*)carve(DD * 2);
  __bf16* Qb  = (__bf16*)carve(BND * 2);
  __bf16* Kb  = (__bf16*)carve(BND * 2);
  __bf16* Vb  = (__bf16*)carve(BND * 2);
  __bf16* Ob  = (__bf16*)carve(BND * 2);
  __bf16* biasT = (__bf16*)carve((size_t)B_ * H_ * N_ * N_ * 2);  // 64 MB
  const bool use_pre = (ws_size >= off);  // deterministic: ws_size is fixed

  cvt_f32_to_bf16<<<(int)((BND + 255) / 256), 256, 0, stream>>>(h, hb, (int)BND);
  cvt_f32_to_bf16<<<(int)((DD + 255) / 256), 256, 0, stream>>>(q_w, qwb, (int)DD);
  cvt_f32_to_bf16<<<(int)((DD + 255) / 256), 256, 0, stream>>>(k_w, kwb, (int)DD);
  cvt_f32_to_bf16<<<(int)((DD + 255) / 256), 256, 0, stream>>>(v_w, vwb, (int)DD);
  cvt_f32_to_bf16<<<(int)((DD + 255) / 256), 256, 0, stream>>>(o_w, owb, (int)DD);

  if (use_pre) {
    rbf_bias_kernel<<<dim3(8, N_, B_), 256, 0, stream>>>(pos, rbf_w, rbf_b, biasT);
  }

  dim3 ggrid((B_ * N_) / 128, D_ / 128);  // (32, 4)
  const float qscale = 0.125f;            // 1/sqrt(HD)
  gemm_bf16_512<__bf16><<<ggrid, 256, 0, stream>>>(hb, qwb, q_b, Qb, qscale);
  gemm_bf16_512<__bf16><<<ggrid, 256, 0, stream>>>(hb, kwb, k_b, Kb, 1.0f);
  gemm_bf16_512<__bf16><<<ggrid, 256, 0, stream>>>(hb, vwb, v_b, Vb, 1.0f);

  dim3 agrid(N_ / 64, H_, B_);  // (16, 8, 4)
  if (use_pre) {
    attn_kernel<false><<<agrid, 128, 0, stream>>>(Qb, Kb, Vb, pos, mask,
                                                  rbf_w, rbf_b, biasT, Ob);
  } else {
    attn_kernel<true><<<agrid, 128, 0, stream>>>(Qb, Kb, Vb, pos, mask,
                                                 rbf_w, rbf_b, biasT, Ob);
  }

  gemm_bf16_512<float><<<ggrid, 256, 0, stream>>>(Ob, owb, o_b, out, 1.0f);
}
